// BertGraphSelfAttention_13297218748514
// MI455X (gfx1250) — compile-verified, matrix operand order checked by reading, not run on hardware
//
#include <hip/hip_runtime.h>
#include <hip/hip_bf16.h>

#define DEV __device__ __forceinline__

typedef __attribute__((ext_vector_type(16))) __bf16 v16bf;
typedef __attribute__((ext_vector_type(8)))  float  v8f;

namespace {

union Frag16 {
  v16bf v;
  uint4 q[2];
  unsigned short s[16];
};

DEV unsigned short f2bf(float f) {
  unsigned int u = __float_as_uint(f);
  u += 0x7FFF + ((u >> 16) & 1);   // round-to-nearest-even
  return (unsigned short)(u >> 16);
}

// Load a 16-element bf16 fragment for the 16x16x32 WMMA layout.
// For A-frags: rowIdx = M-row (lane&15), elements map to K = k0 + kbs + (e&7) + (e&8 ? 16:0).
// For B-frags: rowIdx = N-col when the operand is stored N-major (transposed), same K mapping.
// k0 must already include the per-lane base (kbs = (lane>=16)*8); offsets are 16B aligned.
DEV Frag16 ldfrag(const unsigned short* base, int row, int stride, int k0) {
  Frag16 f;
  const unsigned short* p = base + (size_t)row * stride + k0;
  f.q[0] = *(const uint4*)(p);
  f.q[1] = *(const uint4*)(p + 16);
  return f;
}

DEV v8f wmma_bf16(const Frag16& a, const Frag16& b, v8f c) {
  return __builtin_amdgcn_wmma_f32_16x16x32_bf16(false, a.v, false, b.v,
                                                 (short)0, c, false, false);
}

} // namespace

// ---------------------------------------------------------------------------
// Kernel 1: layernorm tables for the 81 relation labels (padded to 96).
// lnK stored [96][64] (label-major, contiguous in d -> B-frag for qrel GEMM).
// lnVt stored [64][96] (d-major, contiguous in label -> B-frag for wlab GEMM).
// ---------------------------------------------------------------------------
__global__ __launch_bounds__(32) void ln_tables_kernel(
    const float* __restrict__ rel_k, const float* __restrict__ rel_v,
    const float* __restrict__ gk, const float* __restrict__ bkb,
    const float* __restrict__ gv, const float* __restrict__ bvb,
    unsigned short* __restrict__ lnK, unsigned short* __restrict__ lnVt) {
  const int label = blockIdx.x;   // 0..95
  const int lane  = threadIdx.x;  // wave32
  const bool valid = label < 81;

  float k0 = 0.f, k1 = 0.f, v0 = 0.f, v1 = 0.f;
  if (valid) {
    k0 = rel_k[label * 64 + lane];      k1 = rel_k[label * 64 + lane + 32];
    v0 = rel_v[label * 64 + lane];      v1 = rel_v[label * 64 + lane + 32];
  }
  // --- K table ---
  {
    float sum = k0 + k1, sq = k0 * k0 + k1 * k1;
    #pragma unroll
    for (int m = 16; m >= 1; m >>= 1) { sum += __shfl_xor(sum, m); sq += __shfl_xor(sq, m); }
    float mu  = sum * (1.f / 64.f);
    float var = sq * (1.f / 64.f) - mu * mu;
    float rs  = rsqrtf(var + 1e-5f);
    float y0 = valid ? (k0 - mu) * rs * gk[lane]      + bkb[lane]      : 0.f;
    float y1 = valid ? (k1 - mu) * rs * gk[lane + 32] + bkb[lane + 32] : 0.f;
    lnK[label * 64 + lane]      = f2bf(y0);
    lnK[label * 64 + lane + 32] = f2bf(y1);
  }
  // --- V table (stored transposed: [d][label]) ---
  {
    float sum = v0 + v1, sq = v0 * v0 + v1 * v1;
    #pragma unroll
    for (int m = 16; m >= 1; m >>= 1) { sum += __shfl_xor(sum, m); sq += __shfl_xor(sq, m); }
    float mu  = sum * (1.f / 64.f);
    float var = sq * (1.f / 64.f) - mu * mu;
    float rs  = rsqrtf(var + 1e-5f);
    float y0 = valid ? (v0 - mu) * rs * gv[lane]      + bvb[lane]      : 0.f;
    float y1 = valid ? (v1 - mu) * rs * gv[lane + 32] + bvb[lane + 32] : 0.f;
    lnVt[(size_t)lane * 96 + label]        = f2bf(y0);
    lnVt[(size_t)(lane + 32) * 96 + label] = f2bf(y1);
  }
}

// ---------------------------------------------------------------------------
// Kernel 2a: hidden f32 -> bf16
// ---------------------------------------------------------------------------
__global__ void conv_hidden_kernel(const float* __restrict__ x,
                                   unsigned short* __restrict__ y, int n) {
  int i = blockIdx.x * blockDim.x + threadIdx.x;
  if (i < n) y[i] = f2bf(x[i]);
}

// ---------------------------------------------------------------------------
// Kernel 2b: W concat [Wq|Wk|Wv] transposed -> Wt[n][k] bf16 (N-major so the
// WMMA B-fragment's K axis is contiguous).
// ---------------------------------------------------------------------------
__global__ void conv_wt_kernel(const float* __restrict__ Wq,
                               const float* __restrict__ Wk,
                               const float* __restrict__ Wv,
                               unsigned short* __restrict__ Wt) {
  int idx = blockIdx.x * blockDim.x + threadIdx.x;
  if (idx >= 2304 * 768) return;
  int n = idx / 768, k = idx % 768;
  int sel = n / 768;   // always valid: n in [0, 2304)
  int nn  = n % 768;
  const float* W = (sel == 0) ? Wq : (sel == 1) ? Wk : Wv;
  Wt[(size_t)n * 768 + k] = f2bf(W[(size_t)k * 768 + nn]);
}

// ---------------------------------------------------------------------------
// Kernel 3: fused QKV projection GEMM.  Y[4096 x 2304] = X @ [Wq|Wk|Wv] + b.
// One wave per 16x64 output tile, 8 waves per block.
// q,k written [B][H][S][D] bf16; v written transposed [B][H][D][S] bf16 so
// the attention P@V B-fragments load contiguously.
// ---------------------------------------------------------------------------
__global__ __launch_bounds__(256) void qkv_kernel(
    const unsigned short* __restrict__ Xbf, const unsigned short* __restrict__ Wt,
    const float* __restrict__ bq, const float* __restrict__ bk,
    const float* __restrict__ bv,
    unsigned short* __restrict__ qb, unsigned short* __restrict__ kb2,
    unsigned short* __restrict__ vb2) {
  const int lane  = threadIdx.x & 31;
  const int wave  = threadIdx.x >> 5;
  const int colid = lane & 15;
  const int hlf   = lane >> 4;
  const int kbs   = hlf * 8;

  const int t  = blockIdx.x * 8 + wave;      // 9216 tiles: 256 M x 36 N
  const int mt = t / 36, nt = t % 36;
  const int m0 = mt * 16, n0 = nt * 64;

  v8f c0 = {}, c1 = {}, c2 = {}, c3 = {};
  for (int k0 = 0; k0 < 768; k0 += 32) {
    Frag16 a  = ldfrag(Xbf, m0 + colid, 768, k0 + kbs);
    Frag16 b0 = ldfrag(Wt, n0 +      colid, 768, k0 + kbs);
    c0 = wmma_bf16(a, b0, c0);
    Frag16 b1 = ldfrag(Wt, n0 + 16 + colid, 768, k0 + kbs);
    c1 = wmma_bf16(a, b1, c1);
    Frag16 b2 = ldfrag(Wt, n0 + 32 + colid, 768, k0 + kbs);
    c2 = wmma_bf16(a, b2, c2);
    Frag16 b3 = ldfrag(Wt, n0 + 48 + colid, 768, k0 + kbs);
    c3 = wmma_bf16(a, b3, c3);
  }
  #pragma unroll
  for (int j = 0; j < 4; ++j) {
    const int n   = n0 + j * 16 + colid;
    const int sel = n / 768;
    const int nn  = n % 768;
    const int h = nn >> 6, d = nn & 63;
    const float bias = (sel == 0 ? bq : sel == 1 ? bk : bv)[nn];
    v8f cc = (j == 0) ? c0 : (j == 1) ? c1 : (j == 2) ? c2 : c3;
    #pragma unroll
    for (int i = 0; i < 8; ++i) {
      const int m = m0 + i + hlf * 8;          // C-frag row = i + 8*half
      const int batch = m >> 9, s = m & 511;
      unsigned short val = f2bf(cc[i] + bias);
      if (sel == 0)      qb [((size_t)(batch * 12 + h) * 512 + s) * 64 + d] = val;
      else if (sel == 1) kb2[((size_t)(batch * 12 + h) * 512 + s) * 64 + d] = val;
      else               vb2[((size_t)(batch * 12 + h) * 64 + d) * 512 + s] = val;
    }
  }
}

// ---------------------------------------------------------------------------
// Kernel 4: fused graph attention, flash-style, one wave per (b,h,16-q tile).
//   qrel[16x96]  = Q @ lnK^T                        (12 WMMAs, LDS resident)
//   per 32-key chunk: S = Q K^T/8 + qrel[arc] + mask (4 WMMAs + LDS gather)
//   online softmax (shfl cluster-of-16 reductions), wlab label scatter
//   (ds_add_f32), O += P @ V (4 WMMAs via LDS probs transpose)
//   final: ctx = (O + wlab @ lnV) / l               (12 WMMAs)
// ---------------------------------------------------------------------------
__global__ __launch_bounds__(32) void attn_kernel(
    const unsigned short* __restrict__ qbuf, const unsigned short* __restrict__ kbuf,
    const unsigned short* __restrict__ vbuf, const unsigned short* __restrict__ lnK,
    const unsigned short* __restrict__ lnVt, const int* __restrict__ arc,
    const float* __restrict__ amask, float* __restrict__ out) {
  __shared__ float qrel[16][96];
  __shared__ float wlab[16][96];
  __shared__ unsigned short pbf[16][32];
  __shared__ float rsc[16];

  const int lane  = threadIdx.x;
  const int colid = lane & 15;
  const int hlf   = lane >> 4;
  const int kbs   = hlf * 8;

  const int t  = blockIdx.x;            // B*H*32 = 3072
  const int qt = t & 31;
  const int h  = (t >> 5) % 12;
  const int bb = t / 384;
  const int q0 = qt * 16;

  const unsigned short* qh = qbuf + (size_t)(bb * 12 + h) * 512 * 64;
  const unsigned short* kh = kbuf + (size_t)(bb * 12 + h) * 512 * 64;
  const unsigned short* vh = vbuf + (size_t)(bb * 12 + h) * 64 * 512;
  const int* arcb   = arc + (size_t)bb * 512 * 512;
  const float* mskb = amask + (size_t)bb * 512;

  // Q fragments, reused for qrel and all score chunks.
  Frag16 aq0 = ldfrag(qh, q0 + colid, 64, 0  + kbs);
  Frag16 aq1 = ldfrag(qh, q0 + colid, 64, 32 + kbs);

  // qrel = Q @ lnK^T   -> LDS [16][96] f32
  #pragma unroll
  for (int tt = 0; tt < 6; ++tt) {
    Frag16 b0 = ldfrag(lnK, tt * 16 + colid, 64, 0  + kbs);
    Frag16 b1 = ldfrag(lnK, tt * 16 + colid, 64, 32 + kbs);
    v8f c = {};
    c = wmma_bf16(aq0, b0, c);
    c = wmma_bf16(aq1, b1, c);
    #pragma unroll
    for (int i = 0; i < 8; ++i)
      qrel[i + hlf * 8][tt * 16 + colid] = c[i];
  }
  for (int idx = lane; idx < 16 * 96; idx += 32)
    wlab[idx / 96][idx % 96] = 0.f;

  float run_m[8], run_l[8];
  #pragma unroll
  for (int i = 0; i < 8; ++i) { run_m[i] = -1e30f; run_l[i] = 0.f; }
  v8f o0 = {}, o1 = {}, o2 = {}, o3 = {};

  for (int c0 = 0; c0 < 512; c0 += 32) {
    // ---- scores: two 16x16 C tiles over D=64 (2 K-chunks each) ----
    v8f s0 = {}, s1 = {};
    {
      Frag16 bk0 = ldfrag(kh, c0 + colid, 64, 0  + kbs);
      Frag16 bk1 = ldfrag(kh, c0 + colid, 64, 32 + kbs);
      s0 = wmma_bf16(aq0, bk0, s0);
      s0 = wmma_bf16(aq1, bk1, s0);
    }
    {
      Frag16 bk0 = ldfrag(kh, c0 + 16 + colid, 64, 0  + kbs);
      Frag16 bk1 = ldfrag(kh, c0 + 16 + colid, 64, 32 + kbs);
      s1 = wmma_bf16(aq0, bk0, s1);
      s1 = wmma_bf16(aq1, bk1, s1);
    }
    const float m0v = mskb[c0 + colid];
    const float m1v = mskb[c0 + 16 + colid];
    int a0[8], a1[8];
    #pragma unroll
    for (int i = 0; i < 8; ++i) {
      const int row = i + hlf * 8;
      a0[i] = arcb[(size_t)(q0 + row) * 512 + c0 + colid];
      a1[i] = arcb[(size_t)(q0 + row) * 512 + c0 + 16 + colid];
      s0[i] = s0[i] * 0.125f + qrel[row][a0[i]] + m0v;
      s1[i] = s1[i] * 0.125f + qrel[row][a1[i]] + m1v;
    }
    // ---- online softmax: row reductions over the 16-lane clusters ----
    float p0[8], p1[8], sc[8];
    #pragma unroll
    for (int i = 0; i < 8; ++i) {
      float mx = fmaxf(s0[i], s1[i]);
      mx = fmaxf(mx, __shfl_xor(mx, 1));
      mx = fmaxf(mx, __shfl_xor(mx, 2));
      mx = fmaxf(mx, __shfl_xor(mx, 4));
      mx = fmaxf(mx, __shfl_xor(mx, 8));
      const float nm = fmaxf(run_m[i], mx);
      sc[i] = __expf(run_m[i] - nm);
      run_m[i] = nm;
      p0[i] = __expf(s0[i] - nm);
      p1[i] = __expf(s1[i] - nm);
      float rs_ = p0[i] + p1[i];
      rs_ += __shfl_xor(rs_, 1);
      rs_ += __shfl_xor(rs_, 2);
      rs_ += __shfl_xor(rs_, 4);
      rs_ += __shfl_xor(rs_, 8);
      run_l[i] = run_l[i] * sc[i] + rs_;
      o0[i] *= sc[i]; o1[i] *= sc[i]; o2[i] *= sc[i]; o3[i] *= sc[i];
    }
    // broadcast per-row rescale, rescale label bins
    if (colid == 0) {
      #pragma unroll
      for (int i = 0; i < 8; ++i) rsc[i + hlf * 8] = sc[i];
    }
    for (int idx = lane; idx < 16 * 96; idx += 32)
      wlab[idx / 96][idx % 96] *= rsc[idx / 96];
    // scatter probs into label bins (ds_add_f32) + bf16 probs tile for P@V
    #pragma unroll
    for (int i = 0; i < 8; ++i) {
      const int row = i + hlf * 8;
      atomicAdd(&wlab[row][a0[i]], p0[i]);
      atomicAdd(&wlab[row][a1[i]], p1[i]);
      pbf[row][colid]      = f2bf(p0[i]);
      pbf[row][16 + colid] = f2bf(p1[i]);
    }
    // ---- O += P @ V (V stored [D][S] so K axis contiguous) ----
    Frag16 ap;
    ap.q[0] = *(const uint4*)&pbf[colid][kbs];
    ap.q[1] = *(const uint4*)&pbf[colid][kbs + 16];
    {
      Frag16 bv0 = ldfrag(vh,  0 + colid, 512, c0 + kbs);
      o0 = wmma_bf16(ap, bv0, o0);
      Frag16 bv1 = ldfrag(vh, 16 + colid, 512, c0 + kbs);
      o1 = wmma_bf16(ap, bv1, o1);
      Frag16 bv2 = ldfrag(vh, 32 + colid, 512, c0 + kbs);
      o2 = wmma_bf16(ap, bv2, o2);
      Frag16 bv3 = ldfrag(vh, 48 + colid, 512, c0 + kbs);
      o3 = wmma_bf16(ap, bv3, o3);
    }
    if (c0 + 32 < 512)
      __builtin_prefetch(&arcb[(size_t)(q0 + hlf * 8) * 512 + c0 + 32 + colid], 0, 0);
  }

  // ---- ctx_rel = wlab @ lnV  (K = 96 labels, 3 chunks) ----
  Frag16 aw[3];
  #pragma unroll
  for (int kc = 0; kc < 3; ++kc) {
    #pragma unroll
    for (int e = 0; e < 16; ++e) {
      const int lbl = kc * 32 + kbs + (e & 7) + ((e & 8) ? 16 : 0);
      aw[kc].s[e] = f2bf(wlab[colid][lbl]);
    }
  }
  #pragma unroll
  for (int j = 0; j < 4; ++j) {
    v8f rj = {};
    #pragma unroll
    for (int kc = 0; kc < 3; ++kc) {
      Frag16 bw = ldfrag(lnVt, j * 16 + colid, 96, kc * 32 + kbs);
      rj = wmma_bf16(aw[kc], bw, rj);
    }
    v8f oo = (j == 0) ? o0 : (j == 1) ? o1 : (j == 2) ? o2 : o3;
    #pragma unroll
    for (int i = 0; i < 8; ++i) {
      const int row = i + hlf * 8;
      const float val = (oo[i] + rj[i]) / run_l[i];
      out[(size_t)(bb * 512 + q0 + row) * 768 + h * 64 + j * 16 + colid] = val;
    }
  }
}

// ---------------------------------------------------------------------------
extern "C" void kernel_launch(void* const* d_in, const int* in_sizes, int n_in,
                              void* d_out, int out_size, void* d_ws, size_t ws_size,
                              hipStream_t stream) {
  const float* hidden = (const float*)d_in[0];
  const float* amask  = (const float*)d_in[1];
  const int*   arc    = (const int*)d_in[2];
  const float* Wq = (const float*)d_in[3];
  const float* bq = (const float*)d_in[4];
  const float* Wk = (const float*)d_in[5];
  const float* bk = (const float*)d_in[6];
  const float* Wv = (const float*)d_in[7];
  const float* bv = (const float*)d_in[8];
  const float* rel_k = (const float*)d_in[9];
  const float* rel_v = (const float*)d_in[10];
  const float* gk  = (const float*)d_in[11];
  const float* bkb = (const float*)d_in[12];
  const float* gv  = (const float*)d_in[13];
  const float* bvb = (const float*)d_in[14];
  float* out = (float*)d_out;
  (void)in_sizes; (void)n_in; (void)out_size; (void)ws_size;

  char* ws = (char*)d_ws;
  size_t off = 0;
  auto take = [&](size_t bytes) {
    char* p = ws + off;
    off = (off + bytes + 255) & ~(size_t)255;
    return p;
  };
  unsigned short* lnK  = (unsigned short*)take(96 * 64 * 2);
  unsigned short* lnVt = (unsigned short*)take(64 * 96 * 2);
  unsigned short* Xbf  = (unsigned short*)take((size_t)4096 * 768 * 2);
  unsigned short* Wt   = (unsigned short*)take((size_t)2304 * 768 * 2);
  unsigned short* qb   = (unsigned short*)take((size_t)4096 * 768 * 2);
  unsigned short* kbf  = (unsigned short*)take((size_t)4096 * 768 * 2);
  unsigned short* vbf  = (unsigned short*)take((size_t)4096 * 768 * 2);

  ln_tables_kernel<<<96, 32, 0, stream>>>(rel_k, rel_v, gk, bkb, gv, bvb, lnK, lnVt);
  conv_hidden_kernel<<<(4096 * 768 + 255) / 256, 256, 0, stream>>>(hidden, Xbf, 4096 * 768);
  conv_wt_kernel<<<(2304 * 768 + 255) / 256, 256, 0, stream>>>(Wq, Wk, Wv, Wt);
  qkv_kernel<<<1152, 256, 0, stream>>>(Xbf, Wt, bq, bk, bv, qb, kbf, vbf);
  attn_kernel<<<3072, 32, 0, stream>>>(qb, kbf, vbf, lnK, lnVt, arc, amask, out);
}
